// AttributeMultiHeadedAttention_31396210934262
// MI455X (gfx1250) — matrix operand order repaired and checked by reading, not compiled, not run on hardware
//
#include <hip/hip_runtime.h>
#include <hip/hip_bf16.h>

// ---------- problem constants (match reference) ----------
#define B_  4
#define L_  1024
#define A_  8
#define D_  1024
#define H_  16
#define DK_ 64
#define M_  (B_ * L_ * A_)   // 32768 rows for every projection GEMM

// ---------- POD vector types ----------
struct alignas(16) U4 { unsigned int x, y, z, w; };
struct alignas(8)  U2 { unsigned int x, y; };

typedef __attribute__((ext_vector_type(16))) __bf16 v16bf;
typedef __attribute__((ext_vector_type(8)))  float  v8f;
typedef __attribute__((ext_vector_type(4)))  float  f32x4;
typedef __attribute__((ext_vector_type(4)))  __bf16 bf16x4;

__device__ __forceinline__ float asf(unsigned int u) {
    union { unsigned int u; float f; } x; x.u = u; return x.f;
}
__device__ __forceinline__ U2 cvt4_bf16(f32x4 f) {
    union { bf16x4 h; U2 u; } c;
    c.h = __builtin_convertvector(f, bf16x4);   // packed v_cvt_pk_bf16_f32
    return c.u;
}
__device__ __forceinline__ unsigned short f2bf1(float f) {
    union { __bf16 h; unsigned short s; } c; c.h = (__bf16)f; return c.s;
}

// ---------- fp32 -> bf16 bulk convert (weights) ----------
__global__ __launch_bounds__(256)
void cvt_f32_bf16(const float* __restrict__ src, unsigned short* __restrict__ dst) {
    int i = (blockIdx.x * 256 + threadIdx.x) * 4;
    f32x4 f = *(const f32x4*)(src + i);
    *(U2*)(dst + i) = cvt4_bf16(f);
}

// ---------- GEMM: Out[m,n] = sum_k A[m,k] * Wb[n,k] + bias[n] ----------
// 128x128 tile, BK=32, 256 threads = 8 waves (2M x 4N), wave tile 64x32 (4x2 WMMAs).
// A: double-buffered through LDS (fp32->bf16 converted in flight).
// B: fragments loaded DIRECTLY from global (L2-resident weight panel) and
//    register-double-buffered one K-step ahead.
// Main loop is branch-free (all loads unconditional); the last two K-steps are
// peeled so the accumulator def-chain coalesces without per-iteration v_movs.
constexpr int BM = 128, BN = 128, BK = 32;
constexpr int LDST = BK + 8;   // padded LDS row stride (elements)

template <bool A_F32, bool OUT_F32>
__global__ __launch_bounds__(256, 1)
void gemm_bias(const void* __restrict__ Ap, const unsigned short* __restrict__ Wb,
               const float* __restrict__ bias, void* __restrict__ Outp,
               int M, int N, int K)
{
    __shared__ __align__(16) unsigned short As[2][BM * LDST];   // 2 x 10 KB

    const int t     = threadIdx.x;
    const int wave  = t >> 5;
    const int lane  = t & 31;
    const int waveM = wave >> 2;      // 0..1
    const int waveN = wave & 3;       // 0..3
    const int lh    = lane >> 4;      // WMMA half-wave select
    const int l16   = lane & 15;

    const int rowBase = blockIdx.y * BM;
    const int colBase = blockIdx.x * BN;

    const v8f vzero = {0.f, 0.f, 0.f, 0.f, 0.f, 0.f, 0.f, 0.f};
    v8f acc[4][2];
#pragma unroll
    for (int mi = 0; mi < 4; ++mi)
#pragma unroll
        for (int ni = 0; ni < 2; ++ni) acc[mi][ni] = vzero;

    union FragU { v16bf v; U4 q[2]; };

    // staging registers
    f32x4 aF[4];         // A fp32 path (live across WMMA block)
    U4    aH[2];         // A bf16 path
    FragU bA[2], bB[2];  // ping-pong B fragments straight from global

    auto loadA = [&](int k0) {
        if (A_F32) {
            const float* Ag = (const float*)Ap + (size_t)rowBase * K + k0;
#pragma unroll
            for (int i = 0; i < 4; ++i) {
                int idx = t + i * 256;               // 0..1023 float4 chunks
                int r = idx >> 3, c4 = (idx & 7) << 2;
                aF[i] = *(const f32x4*)(Ag + (size_t)r * K + c4);
                if (k0 + BK < K)
                    __builtin_prefetch(Ag + (size_t)r * K + c4 + BK, 0, 1);
            }
        } else {
            const unsigned short* Ag = (const unsigned short*)Ap + (size_t)rowBase * K + k0;
#pragma unroll
            for (int i = 0; i < 2; ++i) {
                int idx = t + i * 256;               // 0..511 16B chunks
                int r = idx >> 2, c8 = (idx & 3) << 3;
                aH[i] = *(const U4*)(Ag + (size_t)r * K + c8);
                if (k0 + BK < K)
                    __builtin_prefetch(Ag + (size_t)r * K + c8 + BK, 0, 1);
            }
        }
    };

    auto storeA = [&](int buf) {
        if (A_F32) {
#pragma unroll
            for (int i = 0; i < 4; ++i) {
                int idx = t + i * 256;
                int r = idx >> 3, c4 = (idx & 7) << 2;
                *(U2*)(&As[buf][r * LDST + c4]) = cvt4_bf16(aF[i]);
            }
        } else {
#pragma unroll
            for (int i = 0; i < 2; ++i) {
                int idx = t + i * 256;
                int r = idx >> 2, c8 = (idx & 3) << 3;
                *(U4*)(&As[buf][r * LDST + c8]) = aH[i];
            }
        }
    };

    // B fragment: row = colBase + waveN*32 + ni*16 + l16, K {k0+lh*8..+7, k0+lh*8+16..+23}
    auto loadB = [&](int k0, FragU* bf) {
        const unsigned short* Wg =
            Wb + (size_t)(colBase + waveN * 32 + l16) * K + k0 + lh * 8;
#pragma unroll
        for (int ni = 0; ni < 2; ++ni) {
            bf[ni].q[0] = *(const U4*)(Wg + (size_t)(ni * 16) * K);
            bf[ni].q[1] = *(const U4*)(Wg + (size_t)(ni * 16) * K + 16);
        }
    };

    auto compute = [&](int buf, const FragU* bf) {
        FragU a[4];
#pragma unroll
        for (int mi = 0; mi < 4; ++mi) {
            // A 16x32 bf16 layout: lane<16 -> K {lh*8..+7, lh*8+16..+23}
            int r = waveM * 64 + mi * 16 + l16;
            const unsigned short* base = &As[buf][r * LDST + lh * 8];
            a[mi].q[0] = *(const U4*)(base);
            a[mi].q[1] = *(const U4*)(base + 16);
        }
#pragma unroll
        for (int mi = 0; mi < 4; ++mi)
#pragma unroll
            for (int ni = 0; ni < 2; ++ni)
                acc[mi][ni] = __builtin_amdgcn_wmma_f32_16x16x32_bf16(
                    false, a[mi].v, false, bf[ni].v, (short)0, acc[mi][ni], false, false);
    };

    // ---- software-pipelined main loop: branch-free body, last 2 steps peeled ----
    loadB(0, bA);          // B for step 0 in flight
    loadA(0);
    storeA(0);
    int cur = 0;
    const int pairs = K / (2 * BK);            // 16 for K=1024
    for (int it = 0; it < pairs - 1; ++it) {   // steps 0..K-2BK-1, all loads unconditional
        int k0 = it * (2 * BK);
        __syncthreads();
        loadB(k0 + BK, bB);
        loadA(k0 + BK);
        compute(cur, bA);
        storeA(cur ^ 1); cur ^= 1;
        __syncthreads();
        loadB(k0 + 2 * BK, bA);
        loadA(k0 + 2 * BK);
        compute(cur, bB);
        storeA(cur ^ 1); cur ^= 1;
    }
    // peeled step K-2BK: consume bA (loaded in last loop half), fetch last tiles
    __syncthreads();
    loadB(K - BK, bB);
    loadA(K - BK);
    compute(cur, bA);
    storeA(cur ^ 1); cur ^= 1;
    // peeled step K-BK: nothing left to fetch
    __syncthreads();
    compute(cur, bB);

    // ---- epilogue: bias add, store ----
#pragma unroll
    for (int mi = 0; mi < 4; ++mi) {
#pragma unroll
        for (int ni = 0; ni < 2; ++ni) {
            int gcol = colBase + waveN * 32 + ni * 16 + l16;
            float bv = bias[gcol];
#pragma unroll
            for (int r = 0; r < 8; ++r) {
                // C/D layout: VGPR r holds M = r + 8*(lane/16), N = lane%16
                int grow  = rowBase + waveM * 64 + mi * 16 + lh * 8 + r;
                float val = acc[mi][ni][r] + bv;
                if (OUT_F32)
                    ((float*)Outp)[(size_t)grow * N + gcol] = val;
                else
                    ((unsigned short*)Outp)[(size_t)grow * N + gcol] = f2bf1(val);
            }
        }
    }
}

// ---------- attention over the attribute axis: one wave per (b,l,h) ----------
__global__ __launch_bounds__(256)
void attn_kernel(const unsigned short* __restrict__ qp,
                 const unsigned short* __restrict__ kp,
                 const unsigned short* __restrict__ vp,
                 unsigned short* __restrict__ xo)   // may alias qp (disjoint per unit)
{
    __shared__ __align__(16) unsigned short sq[8][A_ * DK_];
    __shared__ __align__(16) unsigned short sk[8][A_ * DK_];
    __shared__ __align__(16) unsigned short sv[8][A_ * DK_];
    __shared__ float sp[8][A_][A_];

    const int wave = threadIdx.x >> 5;
    const int lane = threadIdx.x & 31;
    const int unit = blockIdx.x * 8 + wave;          // index over B*L*H
    const int h    = unit & (H_ - 1);
    const int bl   = unit >> 4;                      // b*L + l
    const size_t gbase = (size_t)bl * A_ * D_ + (size_t)h * DK_;

    // cooperative load of q,k,v 8x64 bf16 tiles (16B chunks)
#pragma unroll
    for (int c = 0; c < 2; ++c) {
        int idx = lane + 32 * c;                     // 0..63
        int r   = idx >> 3;
        int off = (idx & 7) << 3;
        size_t g = gbase + (size_t)r * D_ + off;
        *(U4*)(&sq[wave][r * DK_ + off]) = *(const U4*)(qp + g);
        *(U4*)(&sk[wave][r * DK_ + off]) = *(const U4*)(kp + g);
        *(U4*)(&sv[wave][r * DK_ + off]) = *(const U4*)(vp + g);
    }
    __syncthreads();

    // scores (8x8) + softmax across j; bf16 pairs read as u32
#pragma unroll
    for (int p = 0; p < 2; ++p) {
        int i = p * 4 + (lane >> 3);
        int j = lane & 7;
        const unsigned int* qrow = (const unsigned int*)&sq[wave][i * DK_];
        const unsigned int* krow = (const unsigned int*)&sk[wave][j * DK_];
        float s = 0.f;
#pragma unroll
        for (int d = 0; d < DK_ / 2; ++d) {
            unsigned int qa = qrow[d], kb = krow[d];
            s += asf(qa << 16) * asf(kb << 16)
               + asf(qa & 0xffff0000u) * asf(kb & 0xffff0000u);
        }
        s *= 0.125f;                                 // 1/sqrt(64)
        float m = s;
        m = fmaxf(m, __shfl_xor(m, 1, 8));
        m = fmaxf(m, __shfl_xor(m, 2, 8));
        m = fmaxf(m, __shfl_xor(m, 4, 8));
        float e = __expf(s - m);
        float sum = e;
        sum += __shfl_xor(sum, 1, 8);
        sum += __shfl_xor(sum, 2, 8);
        sum += __shfl_xor(sum, 4, 8);
        sp[wave][i][j] = e / sum;
    }
    __syncthreads();

    // out[i, :] = sum_j P[i,j] * V[j, :]
#pragma unroll
    for (int c = 0; c < 2; ++c) {
        int idx = lane + 32 * c;
        int r   = idx >> 3;
        int off = (idx & 7) << 3;
        float o[8] = {0, 0, 0, 0, 0, 0, 0, 0};
#pragma unroll
        for (int j = 0; j < A_; ++j) {
            float pw = sp[wave][r][j];
            const unsigned int* vrow = (const unsigned int*)&sv[wave][j * DK_ + off];
#pragma unroll
            for (int e = 0; e < 4; ++e) {
                unsigned int u = vrow[e];
                o[2 * e]     += pw * asf(u << 16);
                o[2 * e + 1] += pw * asf(u & 0xffff0000u);
            }
        }
        f32x4 lo = { o[0], o[1], o[2], o[3] };
        f32x4 hi = { o[4], o[5], o[6], o[7] };
        union { U2 u[2]; U4 v; } pk;
        pk.u[0] = cvt4_bf16(lo);
        pk.u[1] = cvt4_bf16(hi);
        size_t g = gbase + (size_t)r * D_ + off;
        *(U4*)(xo + g) = pk.v;
    }
}

// ---------- launch ----------
extern "C" void kernel_launch(void* const* d_in, const int* in_sizes, int n_in,
                              void* d_out, int out_size, void* d_ws, size_t ws_size,
                              hipStream_t stream) {
    const float* q_in = (const float*)d_in[0];
    const float* k_in = (const float*)d_in[1];
    const float* v_in = (const float*)d_in[2];
    const float* Wq   = (const float*)d_in[3];
    const float* bq   = (const float*)d_in[4];
    const float* Wk   = (const float*)d_in[5];
    const float* bk   = (const float*)d_in[6];
    const float* Wv   = (const float*)d_in[7];
    const float* bv   = (const float*)d_in[8];
    const float* Wo   = (const float*)d_in[9];
    const float* bo   = (const float*)d_in[10];

    // workspace: q/k/v projections (bf16, 64MB each) + bf16 weights (2MB each)
    unsigned short* qp  = (unsigned short*)d_ws;
    unsigned short* kp  = qp  + (size_t)M_ * D_;
    unsigned short* vp  = kp  + (size_t)M_ * D_;
    unsigned short* wqb = vp  + (size_t)M_ * D_;
    unsigned short* wkb = wqb + (size_t)D_ * D_;
    unsigned short* wvb = wkb + (size_t)D_ * D_;
    unsigned short* wob = wvb + (size_t)D_ * D_;

    const int cvtBlocks = (D_ * D_) / (256 * 4);   // 1024
    cvt_f32_bf16<<<cvtBlocks, 256, 0, stream>>>(Wq, wqb);
    cvt_f32_bf16<<<cvtBlocks, 256, 0, stream>>>(Wk, wkb);
    cvt_f32_bf16<<<cvtBlocks, 256, 0, stream>>>(Wv, wvb);
    cvt_f32_bf16<<<cvtBlocks, 256, 0, stream>>>(Wo, wob);

    dim3 ggrid(D_ / BN, M_ / BM);   // (8, 256)

    gemm_bias<true,  false><<<ggrid, 256, 0, stream>>>(q_in, wqb, bq, qp, M_, D_, D_);
    gemm_bias<true,  false><<<ggrid, 256, 0, stream>>>(k_in, wkb, bk, kp, M_, D_, D_);
    gemm_bias<true,  false><<<ggrid, 256, 0, stream>>>(v_in, wvb, bv, vp, M_, D_, D_);

    attn_kernel<<<(B_ * L_ * H_) / 8, 256, 0, stream>>>(qp, kp, vp, qp);

    gemm_bias<false, true><<<ggrid, 256, 0, stream>>>(qp, wob, bo, (float*)d_out, M_, D_, D_);
}
